// RNNAttention_42803644072461
// MI455X (gfx1250) — matrix operand order repaired and checked by reading, not compiled, not run on hardware
//
#include <hip/hip_runtime.h>

typedef __attribute__((ext_vector_type(16))) __bf16 v16bf;
typedef __attribute__((ext_vector_type(8)))  float  v8f;

union Frag {
    v16bf v;
    uint4 q[2];
    unsigned u32[8];
    unsigned short u[16];
};

__device__ __forceinline__ unsigned short f2bf(float f) {
    // round-to-nearest-even fp32 -> bf16
    unsigned u = __float_as_uint(f);
    unsigned r = (u + 0x7FFFu + ((u >> 16) & 1u)) >> 16;
    return (unsigned short)r;
}

// ---------------------------------------------------------------------------
// elementwise fp32 -> bf16
__global__ void conv_bf16_kernel(const float* __restrict__ src,
                                 unsigned short* __restrict__ dst, int n) {
    int i = blockIdx.x * blockDim.x + threadIdx.x;
    if (i < n) dst[i] = f2bf(src[i]);
}

// fp32 src[R][C] -> bf16 dst[C][R] (transpose+convert), indexed by dst element
__global__ void convT_bf16_kernel(const float* __restrict__ src,
                                  unsigned short* __restrict__ dst, int R, int C) {
    int o = blockIdx.x * blockDim.x + threadIdx.x;
    if (o < R * C) {
        int c = o / R;       // dst row  (= src col)
        int r = o % R;       // dst col  (= src row)
        dst[o] = f2bf(src[(size_t)r * C + c]);
    }
}

// extract v head-transposed: vT[b,h,d,s] = qkv_bf16[b*2048+s, 2048 + h*64 + d]
__global__ void vtrans_kernel(const unsigned short* __restrict__ qkv,
                              unsigned short* __restrict__ vT) {
    int o = blockIdx.x * blockDim.x + threadIdx.x;   // 2^22 elements
    int s = o & 2047;
    int d = (o >> 11) & 63;
    int h = (o >> 17) & 15;
    int b = o >> 21;
    vT[o] = qkv[(unsigned)(b * 2048 + s) * 3072u + 2048u + h * 64u + d];
}

// ---------------------------------------------------------------------------
// C[m,n] = sum_k A[m,k] * Bt[n,k]  (+bias, optional relu)
// A: bf16 row-major [M,K] (lda), Bt: bf16 row-major [N,K] (ldb).
// One wave computes 32(m) x 64(n); block = 4 waves => 128x64 tile.
template <bool RELU, bool OUTF32>
__global__ __launch_bounds__(128)
void gemm_bf16_kernel(const unsigned short* __restrict__ A, int lda,
                      const unsigned short* __restrict__ Bt, int ldb,
                      const float* __restrict__ bias,
                      void* __restrict__ Cp, int ldc, int K) {
    const int lane = threadIdx.x & 31;
    const int wave = threadIdx.x >> 5;
    const int ln = lane & 15, lh = lane >> 4;
    const int m0 = blockIdx.y * 128 + wave * 32;
    const int n0 = blockIdx.x * 64;

    const unsigned arow0 = (unsigned)((m0 + ln) * lda);        // m-tile 0 row
    const unsigned arow1 = (unsigned)((m0 + 16 + ln) * lda);   // m-tile 1 row
    v8f acc[2][4] = {};

    for (int kk = 0; kk < K; kk += 32) {
        // A 16x32 bf16 frags: lane m=ln, K chunks lh*8+[0,8) and 16+lh*8+[0,8)
        Frag a[2];
        a[0].q[0] = *(const uint4*)(A + arow0 + kk + lh * 8);
        a[0].q[1] = *(const uint4*)(A + arow0 + kk + 16 + lh * 8);
        a[1].q[0] = *(const uint4*)(A + arow1 + kk + lh * 8);
        a[1].q[1] = *(const uint4*)(A + arow1 + kk + 16 + lh * 8);
        __builtin_prefetch(A + arow0 + kk + 32, 0, 0);
        __builtin_prefetch(A + arow1 + kk + 32, 0, 0);
#pragma unroll
        for (int t = 0; t < 4; ++t) {
            Frag b;  // B 32x16 frag: lane n=ln, K chunks lh*8+[0,8) and 16+lh*8+[0,8)
            const unsigned boff =
                (unsigned)((n0 + t * 16 + ln) * ldb + kk + lh * 8);
            b.q[0] = *(const uint4*)(Bt + boff);
            b.q[1] = *(const uint4*)(Bt + boff + 16);
#pragma unroll
            for (int mt = 0; mt < 2; ++mt)
                acc[mt][t] = __builtin_amdgcn_wmma_f32_16x16x32_bf16(
                    false, a[mt].v, false, b.v, (short)0, acc[mt][t], false,
                    false);
        }
    }
#pragma unroll
    for (int t = 0; t < 4; ++t) {
        const int n = n0 + t * 16 + ln;
        const float bv = bias[n];
#pragma unroll
        for (int mt = 0; mt < 2; ++mt)
#pragma unroll
            for (int r = 0; r < 8; ++r) {
                const unsigned m = (unsigned)(m0 + mt * 16 + r + 8 * lh);
                float val = acc[mt][t][r] + bv;
                if (RELU) val = val > 0.f ? val : 0.f;
                if (OUTF32)
                    ((float*)Cp)[m * (unsigned)ldc + n] = val;
                else
                    ((unsigned short*)Cp)[m * (unsigned)ldc + n] = f2bf(val);
            }
    }
}

// ---------------------------------------------------------------------------
// Flash-style attention (no softmax): per (b,h, 32-row q-tile):
//   loop over 32-wide k-tiles: w = pq*pk^T ; stage w (f32) in LDS;
//   reload as bf16 A-frag with mask add fused; acc += w * v
__global__ __launch_bounds__(128)
void attn_kernel(const unsigned short* __restrict__ phiq,
                 const unsigned short* __restrict__ phik,
                 const unsigned short* __restrict__ vT,
                 const float* __restrict__ mask,
                 unsigned short* __restrict__ out) {
    __shared__ float wt[4][32 * 32];            // per-wave 32x32 f32 staging
    const int lane = threadIdx.x & 31;
    const int wave = threadIdx.x >> 5;
    const int ln = lane & 15, lh = lane >> 4;
    const int gid = blockIdx.x * 4 + wave;      // 2048 waves total
    const int qt = gid & 63;
    const int h = (gid >> 6) & 15;
    const int b = gid >> 10;
    const int q0 = qt * 32;

    // pq A-fragments: 2 m-tiles x (F=64 -> 2 frags of K=32)
    Frag apq[2][2];
#pragma unroll
    for (int mt = 0; mt < 2; ++mt) {
        const unsigned roff =
            (unsigned)((b * 2048 + q0 + mt * 16 + ln) * 1024 + h * 64);
#pragma unroll
        for (int j = 0; j < 2; ++j) {
            apq[mt][j].q[0] = *(const uint4*)(phiq + roff + j * 32 + lh * 8);
            apq[mt][j].q[1] =
                *(const uint4*)(phiq + roff + j * 32 + 16 + lh * 8);
        }
    }

    v8f acc[2][4] = {};
    float* myw = wt[wave];

    for (int kt = 0; kt < 64; ++kt) {
        const int k0 = kt * 32;
        // --- w tiles: C[q, kk] = sum_f pq[q,f] * pk[kk,f]
        v8f w[2][2] = {};
#pragma unroll
        for (int nt = 0; nt < 2; ++nt) {
            const unsigned koff = (unsigned)((b * 2048 + k0 + nt * 16 + ln) *
                                                 1024 +
                                             h * 64 + lh * 8);
#pragma unroll
            for (int j = 0; j < 2; ++j) {
                Frag bk;
                bk.q[0] = *(const uint4*)(phik + koff + j * 32);
                bk.q[1] = *(const uint4*)(phik + koff + j * 32 + 16);
#pragma unroll
                for (int mt = 0; mt < 2; ++mt)
                    w[mt][nt] = __builtin_amdgcn_wmma_f32_16x16x32_bf16(
                        false, apq[mt][j].v, false, bk.v, (short)0, w[mt][nt],
                        false, false);
            }
        }
        // --- stage raw w (f32) into LDS, row-major 32x32
#pragma unroll
        for (int mt = 0; mt < 2; ++mt)
#pragma unroll
            for (int nt = 0; nt < 2; ++nt)
#pragma unroll
                for (int r = 0; r < 8; ++r)
                    myw[(mt * 16 + r + 8 * lh) * 32 + nt * 16 + ln] =
                        w[mt][nt][r];
        // --- reload as bf16 A-frags with mask add fused in
        Frag aw[2];
#pragma unroll
        for (int mt = 0; mt < 2; ++mt) {
            const float* mrow =
                mask + (unsigned)((q0 + mt * 16 + ln) * 2048 + k0);
            const float* lrow = myw + (mt * 16 + ln) * 32;
#pragma unroll
            for (int vr = 0; vr < 8; ++vr) {
                const int kb = (vr < 4 ? vr * 2 : 16 + (vr - 4) * 2) + lh * 8;
                float2 wv = *(const float2*)(lrow + kb);
                float2 mv = *(const float2*)(mrow + kb);
                aw[mt].u32[vr] = (unsigned)f2bf(wv.x + mv.x) |
                                 ((unsigned)f2bf(wv.y + mv.y) << 16);
            }
        }
        // --- acc[q,d] += w[q,kk] * v[kk,d] ; B frags from vT shared by m-tiles
#pragma unroll
        for (int t = 0; t < 4; ++t) {
            Frag bv;
            const unsigned voff =
                (unsigned)(((b * 16 + h) * 64 + t * 16 + ln) * 2048 + k0 +
                           lh * 8);
            bv.q[0] = *(const uint4*)(vT + voff);
            bv.q[1] = *(const uint4*)(vT + voff + 16);
#pragma unroll
            for (int mt = 0; mt < 2; ++mt)
                acc[mt][t] = __builtin_amdgcn_wmma_f32_16x16x32_bf16(
                    false, aw[mt].v, false, bv.v, (short)0, acc[mt][t], false,
                    false);
        }
    }
    // epilogue: merged-heads layout [b*2048+q, h*64+d] in bf16
#pragma unroll
    for (int mt = 0; mt < 2; ++mt)
#pragma unroll
        for (int t = 0; t < 4; ++t)
#pragma unroll
            for (int r = 0; r < 8; ++r)
                out[(unsigned)((b * 2048 + q0 + mt * 16 + r + 8 * lh) * 1024 +
                               h * 64 + t * 16 + ln)] = f2bf(acc[mt][t][r]);
}

// ---------------------------------------------------------------------------
extern "C" void kernel_launch(void* const* d_in, const int* in_sizes, int n_in,
                              void* d_out, int out_size, void* d_ws,
                              size_t ws_size, hipStream_t stream) {
    const float* x    = (const float*)d_in[0];   // [2,2048,1024]
    const float* mask = (const float*)d_in[1];   // [1,1,2048,2048]
    const float* caw  = (const float*)d_in[2];   // [1024,3072]
    const float* cab  = (const float*)d_in[3];   // [3072]
    const float* phiw = (const float*)d_in[4];   // [1024,1024] (out,in)
    const float* phib = (const float*)d_in[5];   // [1024]
    const float* cpw  = (const float*)d_in[6];   // [1024,1024]
    const float* cpb  = (const float*)d_in[7];   // [1024]
    float* out = (float*)d_out;                  // [2,2048,1024]

    char* ws = (char*)d_ws;
    size_t off = 0;
    auto alloc = [&](size_t elems) -> unsigned short* {
        unsigned short* p = (unsigned short*)(ws + off);
        off = (off + elems * 2 + 255) & ~(size_t)255;
        return p;
    };
    unsigned short* x_bf   = alloc((size_t)4096 * 1024);
    unsigned short* caw_t  = alloc((size_t)3072 * 1024);
    unsigned short* phiw_b = alloc((size_t)1024 * 1024);
    unsigned short* cpw_t  = alloc((size_t)1024 * 1024);
    unsigned short* qkv_bf = alloc((size_t)4096 * 3072);
    unsigned short* phiq   = alloc((size_t)4096 * 1024);
    unsigned short* phik   = alloc((size_t)4096 * 1024);
    unsigned short* vT     = alloc((size_t)4194304);
    unsigned short* attn   = alloc((size_t)4096 * 1024);

    // --- convert / transpose inputs to bf16
    conv_bf16_kernel<<<16384, 256, 0, stream>>>(x, x_bf, 4096 * 1024);
    convT_bf16_kernel<<<12288, 256, 0, stream>>>(caw, caw_t, 1024, 3072);
    conv_bf16_kernel<<<4096, 256, 0, stream>>>(phiw, phiw_b, 1024 * 1024);
    convT_bf16_kernel<<<4096, 256, 0, stream>>>(cpw, cpw_t, 1024, 1024);

    // --- qkv = x @ c_attn_w + b            [4096,3072] bf16
    gemm_bf16_kernel<false, false><<<dim3(48, 32), 128, 0, stream>>>(
        x_bf, 1024, caw_t, 1024, cab, qkv_bf, 3072, 1024);

    // --- phi_q = relu(q @ phi_w^T + b)     [4096,1024] bf16
    gemm_bf16_kernel<true, false><<<dim3(16, 32), 128, 0, stream>>>(
        qkv_bf + 0, 3072, phiw_b, 1024, phib, phiq, 1024, 1024);
    // --- phi_k = relu(k @ phi_w^T + b)
    gemm_bf16_kernel<true, false><<<dim3(16, 32), 128, 0, stream>>>(
        qkv_bf + 1024, 3072, phiw_b, 1024, phib, phik, 1024, 1024);

    // --- vT[b,h,d,s]
    vtrans_kernel<<<16384, 256, 0, stream>>>(qkv_bf, vT);

    // --- attention: out = (phi_q phi_k^T + mask) v      [4096,1024] bf16
    attn_kernel<<<512, 128, 0, stream>>>(phiq, phik, vT, mask, attn);

    // --- final projection to fp32 output
    gemm_bf16_kernel<false, true><<<dim3(16, 32), 128, 0, stream>>>(
        attn, 1024, cpw_t, 1024, cpb, out, 1024, 1024);
}
